// DiscriminatorA_77953656422788
// MI455X (gfx1250) — matrix operand order repaired and checked by reading, not compiled, not run on hardware
//
#include <hip/hip_runtime.h>
#include <hip/hip_bf16.h>
#include <cstdint>

// ---------------------------------------------------------------------------
// Signed-GCN discriminator for gfx1250 (MI455X), bf16 WMMA w/ f32 accumulate.
// One workgroup per (batch, sign). Adjacency slice (160KB f32) is staged into
// LDS once via CDNA5 async copies (global_load_async_to_lds_b128 / ASYNCcnt),
// normalized into a resident bf16 A matrix, and reused across all 4 layers.
// ~260KB static LDS per WG.
// ---------------------------------------------------------------------------

typedef __attribute__((ext_vector_type(16))) __bf16 v16bf;
typedef __attribute__((ext_vector_type(8)))  float  v8f;

#define NN       200     // nodes
#define MPAD     208     // M padded to 13*16
#define KPADN    224     // node-K padded to 7*32
#define ASTR     232     // LDS row stride (halves) for A and Z^T  (464B, /16 ok)
#define NTHREADS 512
#define NWAVES   16
#define D_CC     12800   // 2*N*OUT

// LDS arena offsets in 16-bit units
#define SM_A   0
#define SM_X   (208*232)                       // 48256
#define SM_ZT  (208*232 + 208*264)             // 103168
#define SM_TOT (208*232 + 208*264 + 128*232)   // 132864 halves = 265728 B

__device__ __forceinline__ unsigned short f2bf(float f) {
  unsigned int u = __float_as_uint(f);
  u += 0x7FFFu + ((u >> 16) & 1u);     // round-to-nearest-even
  return (unsigned short)(u >> 16);
}

union V16B { uint4 q[2]; v16bf v; };

// A-operand: halves 0..7 at p, halves 8..15 at p+16 (two 16B LDS loads)
__device__ __forceinline__ v16bf load_a16(const unsigned short* p) {
  V16B t;
  t.q[0] = *(const uint4*)(p);
  t.q[1] = *(const uint4*)(p + 16);
  return t.v;
}
// B-operand: halves 0..7 at p, halves 8..15 at p+16 (two 16B loads)
__device__ __forceinline__ v16bf load_b16(const unsigned short* p) {
  V16B t;
  t.q[0] = *(const uint4*)(p);
  t.q[1] = *(const uint4*)(p + 16);
  return t.v;
}

struct GcnArgs {
  const float* adj;                 // (B,2,200,200)
  const unsigned short* wt[8];      // bf16 W^T per (sign*4+layer): [fo][fi_pad]
  const float* bias[8];             // f32 biases
  float* cc;                        // (B, 200, 64) f32 concat features
};

__global__ __launch_bounds__(NTHREADS)
void gcn_branch_kernel(GcnArgs args) {
  __shared__ unsigned short sm[SM_TOT];
  __shared__ float smD[MPAD];

  const int tid  = threadIdx.x;
  const int lane = tid & 31;
  const int wv   = tid >> 5;
  const int b    = blockIdx.x >> 1;
  const int s    = blockIdx.x & 1;
  const float* adjp = args.adj + (size_t)(b * 2 + s) * (NN * NN);

  const int lrow = lane & 15;             // A row / B col / D col within tile
  const int kbA  = (lane < 16) ? 0 : 8;   // A-operand K sub-offset
  const int kbB  = (lane < 16) ? 0 : 8;   // B-operand K sub-offset
  const int rofs = (lane < 16) ? 0 : 8;   // D row offset per VGPR

  // ---- stage adj (200x200 f32 = 160000B) into X/Z^T region via async DMA ----
  float* stage = (float*)(sm + SM_X);     // 169216B region, used before X/Z live
  {
    unsigned lds0 = (unsigned)(uintptr_t)(void*)stage;   // low 32b = LDS offset
    for (int idx = tid; idx < (NN * NN) / 4; idx += NTHREADS) {  // 16B chunks
      unsigned dst = lds0 + (unsigned)idx * 16u;
      unsigned src = (unsigned)idx * 16u;
      asm volatile("global_load_async_to_lds_b128 %0, %1, %2"
                   :: "v"(dst), "v"(src), "s"(adjp)
                   : "memory");
    }
  }

  // ---- zero A region + degree vector (overlaps with async transfers) ----
  {
    unsigned int* az = (unsigned int*)(sm + SM_A);        // 24128 dwords
    for (int i = tid; i < (208 * 232) / 2; i += NTHREADS) az[i] = 0u;
    if (tid < MPAD) smD[tid] = 0.f;
  }
  asm volatile("s_wait_asynccnt 0" ::: "memory");
  __syncthreads();

  // ---- degree: d_i = rsqrt(1 + sum_j adj[i][j])  (from LDS) ----
  for (int r = wv; r < NN; r += NWAVES) {
    float ssum = 0.f;
    for (int j = lane; j < NN; j += 32) ssum += stage[r * NN + j];
#pragma unroll
    for (int off = 16; off > 0; off >>= 1) ssum += __shfl_down(ssum, off, 32);
    if (lane == 0) smD[r] = rsqrtf(ssum + 1.0f);
  }
  __syncthreads();

  // ---- A_norm = D^-1/2 (adj + I) D^-1/2  -> bf16 LDS, zero-padded ----
  for (int idx = tid; idx < NN * NN; idx += NTHREADS) {
    int i = idx / NN;
    int j = idx - i * NN;
    float v = stage[idx];
    if (i == j) v += 1.0f;
    sm[SM_A + i * ASTR + j] = f2bf(smD[i] * v * smD[j]);
  }
  __syncthreads();

  // ---- stage is dead; zero the Z^T region (K-padding cols must be 0) ----
  {
    unsigned int* zz = (unsigned int*)(sm + SM_ZT);       // 14848 dwords
    for (int i = tid; i < (128 * 232) / 2; i += NTHREADS) zz[i] = 0u;
  }
  __syncthreads();

  // =================== Layer 1:  X = lrelu(A @ W1 + b1) ===================
  // (input features are identity, so Z1 == W1, read from global W1^T)
  {
    const unsigned short* w1t = args.wt[s * 4 + 0];   // [256][224] bf16
    const float*          b1  = args.bias[s * 4 + 0];
    for (int t = wv; t < 13 * 16; t += NWAVES) {
      int mt = t % 13, nt = t / 13;
      const int row = mt * 16 + lrow;
      const int col = nt * 16 + lrow;
      v8f c = {};
#pragma unroll
      for (int kt = 0; kt < 7; ++kt) {
        v16bf a  = load_a16(&sm[SM_A + row * ASTR + kt * 32 + kbA]);
        v16bf bb = load_b16(w1t + col * KPADN + kt * 32 + kbB);
        c = __builtin_amdgcn_wmma_f32_16x16x32_bf16(false, a, false, bb,
                                                    (short)0, c, false, false);
      }
      float bv = b1[col];
#pragma unroll
      for (int r = 0; r < 8; ++r) {
        int orow = mt * 16 + rofs + r;
        float y = c[r] + bv;
        y = (y > 0.f) ? y : 0.2f * y;
        unsigned short bfv = f2bf(y);
        if (orow >= NN) bfv = 0;
        sm[SM_X + orow * 264 + col] = bfv;
      }
    }
  }
  __syncthreads();

  // =================== Layers 2..4 ===================
  const int FI[3] = {256, 128, 64};
  const int FO[3] = {128, 64, 32};
  for (int l = 0; l < 3; ++l) {
    const int fi = FI[l], fo = FO[l];
    const int xin_str  = fi + 8;
    const int xout_str = fo + 8;
    const unsigned short* wt = args.wt[s * 4 + l + 1];   // [fo][fi] bf16
    const float*        bptr = args.bias[s * 4 + l + 1];
    const int ntiles = fo / 16, ktiles = fi / 32;

    // ---- type-1: Z^T = (X @ W)^T ----
    for (int t = wv; t < 13 * ntiles; t += NWAVES) {
      int mt = t % 13, nt = t / 13;
      const int row = mt * 16 + lrow;
      const int col = nt * 16 + lrow;
      v8f c = {};
      for (int kt = 0; kt < ktiles; ++kt) {
        v16bf a  = load_a16(&sm[SM_X + row * xin_str + kt * 32 + kbA]);
        v16bf bb = load_b16(wt + col * fi + kt * 32 + kbB);
        c = __builtin_amdgcn_wmma_f32_16x16x32_bf16(false, a, false, bb,
                                                    (short)0, c, false, false);
      }
#pragma unroll
      for (int r = 0; r < 8; ++r) {
        int orow = mt * 16 + rofs + r;                   // node index
        sm[SM_ZT + col * ASTR + orow] = f2bf(c[r]);      // transposed store
      }
    }
    __syncthreads();

    // ---- type-2: Xnew = act(A @ Z + b)  (last layer -> global cc, f32) ----
    for (int t = wv; t < 13 * ntiles; t += NWAVES) {
      int mt = t % 13, nt = t / 13;
      const int row = mt * 16 + lrow;
      const int col = nt * 16 + lrow;
      v8f c = {};
#pragma unroll
      for (int kt = 0; kt < 7; ++kt) {
        v16bf a  = load_a16(&sm[SM_A + row * ASTR + kt * 32 + kbA]);
        v16bf bb = load_b16(&sm[SM_ZT + col * ASTR + kt * 32 + kbB]);
        c = __builtin_amdgcn_wmma_f32_16x16x32_bf16(false, a, false, bb,
                                                    (short)0, c, false, false);
      }
      float bv = bptr[col];
      if (l < 2) {
#pragma unroll
        for (int r = 0; r < 8; ++r) {
          int orow = mt * 16 + rofs + r;
          float y = c[r] + bv;
          y = (y > 0.f) ? y : 0.2f * y;
          unsigned short bfv = f2bf(y);
          if (orow >= NN) bfv = 0;
          sm[SM_X + orow * xout_str + col] = bfv;
        }
      } else {
        float* ccp = args.cc + (size_t)b * D_CC + s * 32;
#pragma unroll
        for (int r = 0; r < 8; ++r) {
          int orow = mt * 16 + rofs + r;
          if (orow < NN) ccp[orow * 64 + col] = c[r] + bv;
        }
      }
    }
    __syncthreads();
  }
}

// ---- weight pre-transpose + f32->bf16 convert: WT[n][k] = bf16(W[k][n]) ----
__global__ void wt_convert_kernel(const float* W, unsigned short* WT,
                                  int fi, int fo, int fi_pad) {
  int idx = blockIdx.x * blockDim.x + threadIdx.x;
  if (idx >= fo * fi_pad) return;
  int n = idx / fi_pad;
  int k = idx - n * fi_pad;
  WT[idx] = (k < fi) ? f2bf(W[(size_t)k * fo + n]) : (unsigned short)0;
}

// ---- head: out1 = cc@Wl1+bl1 ; out2 = softmax(cc@Wl2+bl2) ----
__global__ __launch_bounds__(256)
void head_kernel(const float* cc, const float* Wl1, const float* bl1,
                 const float* Wl2, const float* bl2, float* out) {
  __shared__ float r0[256], r1[256], r2[256];
  const int b = blockIdx.x, tid = threadIdx.x;
  const float* c = cc + (size_t)b * D_CC;
  float s0 = 0.f, s1 = 0.f, s2 = 0.f;
  for (int i = tid; i < D_CC; i += 256) {
    float v = c[i];
    s0 += v * Wl1[i];
    s1 += v * Wl2[i * 2 + 0];
    s2 += v * Wl2[i * 2 + 1];
  }
  r0[tid] = s0; r1[tid] = s1; r2[tid] = s2;
  __syncthreads();
  for (int st = 128; st > 0; st >>= 1) {
    if (tid < st) { r0[tid] += r0[tid + st]; r1[tid] += r1[tid + st]; r2[tid] += r2[tid + st]; }
    __syncthreads();
  }
  if (tid == 0) {
    out[b] = r0[0] + bl1[0];
    float l0 = r1[0] + bl2[0], l1 = r2[0] + bl2[1];
    float m  = fmaxf(l0, l1);
    float e0 = __expf(l0 - m), e1 = __expf(l1 - m);
    float inv = 1.0f / (e0 + e1);
    out[512 + b * 2 + 0] = e0 * inv;
    out[512 + b * 2 + 1] = e1 * inv;
  }
}

extern "C" void kernel_launch(void* const* d_in, const int* in_sizes, int n_in,
                              void* d_out, int out_size, void* d_ws, size_t ws_size,
                              hipStream_t stream) {
  (void)in_sizes; (void)n_in; (void)out_size; (void)ws_size;
  const float* adj = (const float*)d_in[0];

  static const int FI[4]  = {200, 256, 128, 64};
  static const int FIP[4] = {224, 256, 128, 64};  // fi padded (layer1 -> 224)
  static const int FO[4]  = {256, 128, 64, 32};

  char*  ws = (char*)d_ws;
  float* cc = (float*)ws;
  size_t off = (size_t)512 * D_CC * sizeof(float);   // 26,214,400 B (16B aligned)

  GcnArgs args;
  args.adj = adj;
  args.cc  = cc;

  for (int s = 0; s < 2; ++s) {
    for (int l = 0; l < 4; ++l) {
      unsigned short* wtp = (unsigned short*)(ws + off);
      off += (size_t)FO[l] * FIP[l] * sizeof(unsigned short);
      const float* W = (const float*)d_in[1 + s * 8 + l * 2];
      int total = FO[l] * FIP[l];
      wt_convert_kernel<<<(total + 255) / 256, 256, 0, stream>>>(
          W, wtp, FI[l], FO[l], FIP[l]);
      args.wt[s * 4 + l]   = wtp;
      args.bias[s * 4 + l] = (const float*)d_in[2 + s * 8 + l * 2];
    }
  }

  gcn_branch_kernel<<<1024, NTHREADS, 0, stream>>>(args);

  head_kernel<<<512, 256, 0, stream>>>(cc,
      (const float*)d_in[17], (const float*)d_in[18],
      (const float*)d_in[19], (const float*)d_in[20],
      (float*)d_out);
}
